// MPNN_12867722019087
// MI455X (gfx1250) — matrix operand order repaired and checked by reading, not compiled, not run on hardware
//
#include <hip/hip_runtime.h>
#include <hip/hip_bf16.h>

#define DD   64
#define DINP 7
#define GG   64

typedef __attribute__((ext_vector_type(16))) __bf16   v16bf;
typedef __attribute__((ext_vector_type(8)))  float    v8f;
typedef __attribute__((ext_vector_type(4)))  unsigned v4u;
typedef __attribute__((ext_vector_type(8)))  int      v8i;
typedef __attribute__((ext_vector_type(4)))  int      v4i;

__device__ __forceinline__ v8f wmma_bf16(v16bf a, v16bf b, v8f c) {
  // D = A(16x32 bf16) * B(32x16 bf16) + C(16x16 f32)
  return __builtin_amdgcn_wmma_f32_16x16x32_bf16(false, a, false, b, (short)0, c, false, false);
}

__device__ __forceinline__ void atomAddF(float* p, float v) {
  __hip_atomic_fetch_add(p, v, __ATOMIC_RELAXED, __HIP_MEMORY_SCOPE_AGENT);
}

// Workgroup-relative LDS byte offset of a __shared__ object (addrspace(3) ptr-to-int).
__device__ __forceinline__ unsigned lds_offset_of(const void* p) {
  return (unsigned)(unsigned long long)(const __attribute__((address_space(3))) char*)p;
}

// Issue a 1-D TDM copy: nbytes (multiple of 4) from 4-aligned global -> LDS offset.
// D# built per CDNA5 ISA 8.3/8.4; groups 2/3 zero (<=2D tensor). Tracked by TENSORcnt.
__device__ __forceinline__ void tdm_load_to_lds(const void* gsrc, unsigned lds_byte_off,
                                                unsigned nbytes) {
  const unsigned long long ga = (unsigned long long)gsrc;
  const unsigned units = nbytes >> 2;         // data_size = 4B units
  v4u g0;
  g0.x = 1u;                                  // count=1, user desc, no gather
  g0.y = lds_byte_off;                        // lds_addr
  g0.z = (unsigned)(ga & 0xffffffffu);        // global_addr[31:0]
  g0.w = (unsigned)((ga >> 32) & 0x1ffffffu)  // global_addr[56:32]
         | (2u << 30);                        // type = 2 ("image")
  v8i g1;
  g1[0] = (int)(2u << 16);                    // workgroup_mask=0, data_size=2 (4B)
  g1[1] = (int)((units & 0xffffu) << 16);     // tensor_dim0[15:0] in bits 63:48
  g1[2] = (int)(((units >> 16) & 0xffffu)     // tensor_dim0[31:16]
         | (1u << 16));                       // tensor_dim1 = 1
  g1[3] = (int)((units & 0xffffu) << 16);     // tile_dim0 (bits 127:112)
  g1[4] = 1;                                  // tile_dim1 = 1, tile_dim2 = 0
  g1[5] = (int)units;                         // tensor_dim0_stride[31:0]
  g1[6] = 0;                                  // stride0 hi, stride1 lo
  g1[7] = 0;
  v4i z4 = {};
  v8i z8 = {};
  __builtin_amdgcn_tensor_load_to_lds(g0, g1, z4, z4, z8, 0);
}

// ---- A fragment (16x32 bf16) from f32 row-major source, ISA 16-bit A layout:
//   element i of lane L -> K = kbase + (i&7) + 16*(i>>3) + 8*(L>>4), M = L&15
__device__ __forceinline__ v16bf a_frag_f32(const float* src, int ld, long row,
                                            int kbase, int lane, float scale) {
  const float* p = src + row * (long)ld + kbase + ((lane >> 4) << 3);
  v16bf a;
#pragma unroll
  for (int i = 0; i < 8; ++i) a[i]     = (__bf16)(p[i] * scale);
#pragma unroll
  for (int i = 0; i < 8; ++i) a[i + 8] = (__bf16)(p[16 + i] * scale);
  return a;
}

// A fragment where K==63 is replaced by degree (concat([node_edge_mean, deg]))
__device__ __forceinline__ v16bf a_frag_aggdeg(const float* agg, const float* deg,
                                               long row, int kbase, int lane, float inv) {
  const int kb = kbase + ((lane >> 4) << 3);
  v16bf a;
#pragma unroll
  for (int i = 0; i < 16; ++i) {
    const int k = kb + (i & 7) + ((i >> 3) << 4);
    const float v = (k == DD - 1) ? deg[row] : agg[row * (long)DD + k] * inv;
    a[i] = (__bf16)v;
  }
  return a;
}

// ---- B fragment: pre-packed so each lane's 16 bf16 are contiguous (32B load)
__device__ __forceinline__ v16bf b_frag(const __bf16* packed, int frag, int lane) {
  return *(const v16bf*)(packed + ((((long)frag * 32) + lane) << 4));
}

// ---- C/D store with ReLU; layout: VGPR r -> M = r + 8*(L>>4), N = L&15.
// Wave-uniform full-tile fast path keeps the hot loop branchless.
__device__ __forceinline__ void store_d_relu(float* dst, long n0, int ncol0, int lane,
                                             v8f c, long nmax) {
  const int col = ncol0 + (lane & 15);
  const long rbase = n0 + ((lane >> 4) << 3);
  float* p = dst + rbase * (long)DD + col;
  if (n0 + 16 <= nmax) {                 // uniform: full tile, no per-row guards
#pragma unroll
    for (int r = 0; r < 8; ++r) p[r * DD] = fmaxf(c[r], 0.0f);
  } else {
#pragma unroll
    for (int r = 0; r < 8; ++r)
      if (rbase + r < nmax) p[r * DD] = fmaxf(c[r], 0.0f);
  }
}

// ============================ kernels ============================

__global__ void fill_zero(float* p, long n) {
  long t = (long)blockIdx.x * blockDim.x + threadIdx.x;
  if (t < n) p[t] = 0.0f;
}

// Pack f32 weight matrix W[K x 64] (row-major) into bf16 WMMA B-fragment layout.
// frag = kc*4 + nt ; element i of lane L -> W[kc*32 + 8*(L>>4) + (i&7) + 16*(i>>3)][nt*16 + (L&15)]
__global__ void pack_weights(const float* __restrict__ W, __bf16* __restrict__ out, int KC) {
  const int t = blockIdx.x * blockDim.x + threadIdx.x;
  const int total = KC * 4 * 32 * 16;
  if (t >= total) return;
  const int i = t & 15;
  const int lane = (t >> 4) & 31;
  const int frag = t >> 9;
  const int kc = frag >> 2, nt = frag & 3;
  const int k = kc * 32 + ((lane >> 4) << 3) + (i & 7) + ((i >> 3) << 4);
  const int n = nt * 16 + (lane & 15);
  out[t] = (__bf16)W[(long)k * DD + n];
}

// x_emb = relu(x @ W_embed_node), K = 7 (tiny, VALU)
__global__ void node_embed_kernel(const float* __restrict__ x, const float* __restrict__ Wn,
                                  float* __restrict__ xe, int nnodes) {
  long t = (long)blockIdx.x * blockDim.x + threadIdx.x;
  if (t >= (long)nnodes * DD) return;
  const long n = t >> 6;
  const int d = (int)(t & 63);
  float acc = 0.0f;
#pragma unroll
  for (int k = 0; k < DINP; ++k) acc += x[n * DINP + k] * Wn[k * DD + d];
  xe[t] = fmaxf(acc, 0.0f);
}

// per edge: relu([edge_attr, x[col]] @ W_ne) scatter-add to agg[row][0:63];
// also row counts (reused by every seg-mean over `row`) and col degree.
__global__ void edge_embed_kernel(const float* __restrict__ x, const int* __restrict__ ei,
                                  const float* __restrict__ ea, const float* __restrict__ Wne,
                                  float* __restrict__ agg, float* __restrict__ rowcnt,
                                  float* __restrict__ deg, int ne) {
  long e = (long)blockIdx.x * blockDim.x + threadIdx.x;
  if (e >= ne) return;
  const int r = ei[e];
  const int c = ei[(long)ne + e];
  float f[DINP + 1];
  f[0] = ea[e];
#pragma unroll
  for (int k = 0; k < DINP; ++k) f[k + 1] = x[(long)c * DINP + k];
  for (int d = 0; d < DD - 1; ++d) {
    float v = 0.0f;
#pragma unroll
    for (int k = 0; k < DINP + 1; ++k) v += f[k] * Wne[k * (DD - 1) + d];
    atomAddF(&agg[(long)r * DD + d], fmaxf(v, 0.0f));
  }
  atomAddF(&rowcnt[r], 1.0f);
  atomAddF(&deg[c], 1.0f);
}

// x_agg_emb = relu(concat([node_edge_mean, deg]) @ W_embed_agg) — WMMA, K=64
__global__ void agg_embed_kernel(const float* __restrict__ agg, const float* __restrict__ rowcnt,
                                 const float* __restrict__ deg, const __bf16* __restrict__ pagg,
                                 float* __restrict__ xagge, int nnodes) {
  const int lane = threadIdx.x & 31;
  const long n0 = (long)blockIdx.x * 16;
  long rowA = n0 + (lane & 15);
  if (rowA >= nnodes) rowA = nnodes - 1;
  const float inv = 1.0f / fmaxf(rowcnt[rowA], 1.0f);
  __builtin_prefetch(pagg, 0, 3);
  const v16bf a0 = a_frag_aggdeg(agg, deg, rowA, 0, lane, inv);
  const v16bf a1 = a_frag_aggdeg(agg, deg, rowA, 32, lane, inv);
#pragma unroll
  for (int nt = 0; nt < 4; ++nt) {
    v8f c = {};
    c = wmma_bf16(a0, b_frag(pagg, 0 * 4 + nt, lane), c);
    c = wmma_bf16(a1, b_frag(pagg, 1 * 4 + nt, lane), c);
    store_d_relu(xagge, n0, nt * 16, lane, c, nnodes);
  }
}

// x_agg accumulation: agg[row] += edge_attr * x_emb[col]  (16 lanes/edge, float4)
__global__ void gather_scatter_kernel(const float* __restrict__ xe, const int* __restrict__ ei,
                                      const float* __restrict__ ea, float* __restrict__ agg, int ne) {
  long t = (long)blockIdx.x * blockDim.x + threadIdx.x;
  const long e = t >> 4;
  if (e >= ne) return;
  const int s = (int)(t & 15);
  const int r = ei[e];
  const int c = ei[(long)ne + e];
  const float a = ea[e];
  const float4 v = *(const float4*)(xe + (long)c * DD + s * 4);
  float* dst = &agg[(long)r * DD + s * 4];
  atomAddF(dst + 0, a * v.x);
  atomAddF(dst + 1, a * v.y);
  atomAddF(dst + 2, a * v.z);
  atomAddF(dst + 3, a * v.w);
}

// Fused per-layer kernel, 8 waves / 128 nodes per block.
//   Phase 1: TDM-stage packed Wmsg into LDS; m = relu([x_agg_mean | x_agg_emb] @ Wmsg) -> LDS
//   Phase 2: TDM-stage packed Wupd into LDS; x' = relu([x_emb | m] @ Wupd) -> global
// Weights are read from L2 once per 128 nodes instead of once per 16.
__global__ void layer_kernel(const float* __restrict__ agg, const float* __restrict__ rowcnt,
                             const float* __restrict__ xagge, const float* __restrict__ xe,
                             const __bf16* __restrict__ pmsg, const __bf16* __restrict__ pupd,
                             float* __restrict__ xout, int nnodes) {
  __shared__ __bf16 wbuf[16 * 32 * 16];        // 8192 bf16 = 16 KB (one packed matrix)
  __shared__ float  mtile[8 * 16 * DD];        // 32 KB (per-wave m tiles)

  const int lane = threadIdx.x & 31;
  const int wid  = threadIdx.x >> 5;
  const long n0  = ((long)blockIdx.x * 8 + wid) * 16;
  long rowA = n0 + (lane & 15);
  if (rowA >= nnodes) rowA = nnodes - 1;
  const float inv = 1.0f / fmaxf(rowcnt[rowA], 1.0f);
  float* mt = &mtile[wid * 16 * DD];

  // ---- stage Wmsg via Tensor Data Mover (wave 0 only; TENSORcnt tracked) ----
  if (threadIdx.x < 32) {
    tdm_load_to_lds(pmsg, lds_offset_of(wbuf), 16 * 32 * 16 * 2);
    __builtin_amdgcn_s_wait_tensorcnt(0);
  }
  __syncthreads();

  // ---- message GEMM, K=128 (kc0/1: normalized x_agg, kc2/3: x_agg_emb) ----
  {
    const v16bf a0 = a_frag_f32(agg,   DD, rowA, 0,  lane, inv);
    const v16bf a1 = a_frag_f32(agg,   DD, rowA, 32, lane, inv);
    const v16bf a2 = a_frag_f32(xagge, DD, rowA, 0,  lane, 1.0f);
    const v16bf a3 = a_frag_f32(xagge, DD, rowA, 32, lane, 1.0f);
#pragma unroll
    for (int nt = 0; nt < 4; ++nt) {
      v8f c = {};
      c = wmma_bf16(a0, b_frag(wbuf, 0 * 4 + nt, lane), c);
      c = wmma_bf16(a1, b_frag(wbuf, 1 * 4 + nt, lane), c);
      c = wmma_bf16(a2, b_frag(wbuf, 2 * 4 + nt, lane), c);
      c = wmma_bf16(a3, b_frag(wbuf, 3 * 4 + nt, lane), c);
      const int col = nt * 16 + (lane & 15);
      const int rb = ((lane >> 4) << 3);
#pragma unroll
      for (int r = 0; r < 8; ++r) mt[(rb + r) * DD + col] = fmaxf(c[r], 0.0f);  // ReLU -> LDS
    }
  }
  __syncthreads();                      // all waves done reading Wmsg from wbuf

  // ---- stage Wupd over the same LDS buffer ----
  if (threadIdx.x < 32) {
    tdm_load_to_lds(pupd, lds_offset_of(wbuf), 16 * 32 * 16 * 2);
    __builtin_amdgcn_s_wait_tensorcnt(0);
  }
  __syncthreads();

  // ---- update GEMM, K=128 (kc0/1: x_emb from global, kc2/3: m from LDS) ----
  {
    const v16bf u0 = a_frag_f32(xe, DD, rowA, 0,  lane, 1.0f);
    const v16bf u1 = a_frag_f32(xe, DD, rowA, 32, lane, 1.0f);
    const v16bf u2 = a_frag_f32(mt, DD, (long)(lane & 15), 0,  lane, 1.0f);
    const v16bf u3 = a_frag_f32(mt, DD, (long)(lane & 15), 32, lane, 1.0f);
#pragma unroll
    for (int nt = 0; nt < 4; ++nt) {
      v8f c = {};
      c = wmma_bf16(u0, b_frag(wbuf, 0 * 4 + nt, lane), c);
      c = wmma_bf16(u1, b_frag(wbuf, 1 * 4 + nt, lane), c);
      c = wmma_bf16(u2, b_frag(wbuf, 2 * 4 + nt, lane), c);
      c = wmma_bf16(u3, b_frag(wbuf, 3 * 4 + nt, lane), c);
      store_d_relu(xout, n0, nt * 16, lane, c, nnodes);
    }
  }
}

// global mean pool accumulation
__global__ void pool_kernel(const float* __restrict__ xe, const int* __restrict__ batch,
                            float* __restrict__ gsum, float* __restrict__ gcnt, int nnodes) {
  long t = (long)blockIdx.x * blockDim.x + threadIdx.x;
  if (t >= (long)nnodes * DD) return;
  const long n = t >> 6;
  const int d = (int)(t & 63);
  const int b = batch[n];
  atomAddF(&gsum[(long)b * DD + d], xe[t]);
  if (d == 0) atomAddF(&gcnt[b], 1.0f);
}

// g = relu((gsum/cnt) @ W_agg_nodes), 64x64 — tiny, VALU
__global__ void graph_mlp_kernel(const float* __restrict__ gsum, const float* __restrict__ gcnt,
                                 const float* __restrict__ Wagg, float* __restrict__ g) {
  const int t = blockIdx.x * blockDim.x + threadIdx.x;
  if (t >= GG * DD) return;
  const int i = t >> 6, j = t & 63;
  const float inv = 1.0f / fmaxf(gcnt[i], 1.0f);
  float acc = 0.0f;
  for (int k = 0; k < DD; ++k) acc += gsum[i * DD + k] * inv * Wagg[k * DD + j];
  g[t] = fmaxf(acc, 0.0f);
}

// q = concat([g[batch], x_emb]) @ W_readout + b
__global__ void readout_kernel(const float* __restrict__ g, const float* __restrict__ xe,
                               const int* __restrict__ batch, const float* __restrict__ Wr,
                               const float* __restrict__ br, float* __restrict__ out, int nnodes) {
  long n = (long)blockIdx.x * blockDim.x + threadIdx.x;
  if (n >= nnodes) return;
  const int b = batch[n];
  float q = br[0];
  for (int k = 0; k < DD; ++k) q += g[(long)b * DD + k] * Wr[k];
  for (int k = 0; k < DD; ++k) q += xe[n * (long)DD + k] * Wr[DD + k];
  out[n] = q;
}

// ============================ launch ============================

extern "C" void kernel_launch(void* const* d_in, const int* in_sizes, int n_in,
                              void* d_out, int out_size, void* d_ws, size_t ws_size,
                              hipStream_t stream) {
  const float* x     = (const float*)d_in[0];
  const int*   ei    = (const int*)d_in[1];
  const float* ea    = (const float*)d_in[2];
  const int*   batch = (const int*)d_in[3];
  const float* Wnode = (const float*)d_in[4];
  const float* Wne   = (const float*)d_in[5];
  const float* Waggm = (const float*)d_in[6];
  const float* Wmsg  = (const float*)d_in[7];
  const float* Wupd  = (const float*)d_in[8];
  const float* Waggn = (const float*)d_in[9];
  const float* Wread = (const float*)d_in[10];
  const float* bread = (const float*)d_in[11];

  const int N = in_sizes[0] / DINP;
  const int E = in_sizes[2];                 // edge_attr is (E,1)
  const int L = in_sizes[7] / (2 * DD * DD); // W_msg is (L, 2D, D)

  char* ws = (char*)d_ws;
  size_t off = 0;
  auto carve = [&](size_t bytes) -> void* {
    void* p = ws + off;
    off = (off + bytes + 255) & ~(size_t)255;
    return p;
  };
  float*  xe0    = (float*)carve((size_t)N * DD * 4);
  float*  xe1    = (float*)carve((size_t)N * DD * 4);
  float*  xagge  = (float*)carve((size_t)N * DD * 4);
  float*  agg    = (float*)carve((size_t)N * DD * 4);
  float*  rowcnt = (float*)carve((size_t)N * 4);
  float*  deg    = (float*)carve((size_t)N * 4);
  float*  gsum   = (float*)carve((size_t)GG * DD * 4);
  float*  gcnt   = (float*)carve((size_t)GG * 4);
  float*  g      = (float*)carve((size_t)GG * DD * 4);
  __bf16* pagg   = (__bf16*)carve((size_t)2 * 4 * 512 * 2);            // K=64  -> 2 kc
  __bf16* pmsg   = (__bf16*)carve((size_t)L * 4 * 4 * 512 * 2);        // K=128 -> 4 kc
  __bf16* pupd   = (__bf16*)carve((size_t)L * 4 * 4 * 512 * 2);

  const int TB = 256;
  auto blocks = [](long n, int tb) { return (unsigned)((n + tb - 1) / tb); };

  // zero accumulators
  fill_zero<<<blocks((long)N * DD, TB), TB, 0, stream>>>(agg, (long)N * DD);
  fill_zero<<<blocks(N, TB), TB, 0, stream>>>(rowcnt, N);
  fill_zero<<<blocks(N, TB), TB, 0, stream>>>(deg, N);
  fill_zero<<<blocks(GG * DD, TB), TB, 0, stream>>>(gsum, GG * DD);
  fill_zero<<<blocks(GG, TB), TB, 0, stream>>>(gcnt, GG);

  // pack weights to bf16 WMMA B-layout
  pack_weights<<<blocks(2 * 4 * 512, TB), TB, 0, stream>>>(Waggm, pagg, 2);
  for (int l = 0; l < L; ++l) {
    pack_weights<<<blocks(4 * 4 * 512, TB), TB, 0, stream>>>(Wmsg + (size_t)l * 2 * DD * DD,
                                                             pmsg + (size_t)l * 4 * 4 * 512, 4);
    pack_weights<<<blocks(4 * 4 * 512, TB), TB, 0, stream>>>(Wupd + (size_t)l * 2 * DD * DD,
                                                             pupd + (size_t)l * 4 * 4 * 512, 4);
  }

  // node embed
  node_embed_kernel<<<blocks((long)N * DD, TB), TB, 0, stream>>>(x, Wnode, xe0, N);

  // edge embed + counts
  edge_embed_kernel<<<blocks(E, TB), TB, 0, stream>>>(x, ei, ea, Wne, agg, rowcnt, deg, E);

  // x_agg_emb (WMMA, 1 wave per 16-node tile)
  const unsigned tiles16 = (unsigned)((N + 15) / 16);
  agg_embed_kernel<<<tiles16, 32, 0, stream>>>(agg, rowcnt, deg, pagg, xagge, N);

  // message-passing layers (8 waves per block, TDM-staged weights)
  const unsigned tiles128 = (unsigned)(((long)N + 127) / 128);
  float* xcur = xe0;
  float* xnxt = xe1;
  for (int l = 0; l < L; ++l) {
    fill_zero<<<blocks((long)N * DD, TB), TB, 0, stream>>>(agg, (long)N * DD);
    gather_scatter_kernel<<<blocks((long)E * 16, TB), TB, 0, stream>>>(xcur, ei, ea, agg, E);
    layer_kernel<<<tiles128, 256, 0, stream>>>(agg, rowcnt, xagge, xcur,
                                               pmsg + (size_t)l * 4 * 4 * 512,
                                               pupd + (size_t)l * 4 * 4 * 512, xnxt, N);
    float* t = xcur; xcur = xnxt; xnxt = t;
  }

  // readout
  pool_kernel<<<blocks((long)N * DD, TB), TB, 0, stream>>>(xcur, batch, gsum, gcnt, N);
  graph_mlp_kernel<<<blocks(GG * DD, TB), TB, 0, stream>>>(gsum, gcnt, Waggn, g);
  readout_kernel<<<blocks(N, TB), TB, 0, stream>>>(g, xcur, batch, Wread, bread, (float*)d_out, N);
}
